// OnlineTripletLoss_35442070127217
// MI455X (gfx1250) — matrix operand order repaired and checked by reading, not compiled, not run on hardware
//
#include <hip/hip_runtime.h>
#include <hip/hip_bf16.h>

typedef _Float16 v8h  __attribute__((ext_vector_type(8)));
typedef _Float16 v16h __attribute__((ext_vector_type(16)));
typedef float    v8f  __attribute__((ext_vector_type(8)));

#define TL_B 8192
#define TL_D 128
#define TL_MARGIN 1.0f
#define TL_EPS 1e-8f
#define COL_SPLITS 8
#define COLS_PER_SPLIT (TL_B / COL_SPLITS)   // 1024
#define CTILE 32                             // columns per LDS tile
#define ROWS_PER_WG 128                      // 8 waves * 16 rows
#define LDS_STRIDE 136                       // 128 + 8 f16 pad -> conflict-free b128 reads

// ---------------------------------------------------------------------------
// Kernel 1: row squared norms + init of pos2/neg2 accumulator arrays in ws.
// ---------------------------------------------------------------------------
__global__ void triplet_init_kernel(const float* __restrict__ E,
                                    float* __restrict__ sq,
                                    float* __restrict__ pos2,
                                    float* __restrict__ neg2) {
  int i = blockIdx.x * blockDim.x + threadIdx.x;
  if (i >= TL_B) return;
  const float4* row = (const float4*)(E + (size_t)i * TL_D);
  float s = 0.0f;
#pragma unroll
  for (int k = 0; k < TL_D / 4; ++k) {
    float4 v = row[k];
    s += v.x * v.x + v.y * v.y + v.z * v.z + v.w * v.w;
  }
  sq[i]   = s;
  pos2[i] = -__builtin_inff();
  neg2[i] =  __builtin_inff();
}

// ---------------------------------------------------------------------------
// Kernel 2: tiled Gram via split-f16 WMMA + branch-free batch-hard mining.
// Grid: (COL_SPLITS, TL_B / ROWS_PER_WG), 256 threads (8 wave32).
// Each wave holds a 16x128 A tile (hi/lo f16) in VGPRs and computes two
// 16x16 C subtiles per 32-column LDS tile (24 WMMAs between barriers).
// ---------------------------------------------------------------------------
__global__ void __launch_bounds__(256)
triplet_main_kernel(const float* __restrict__ E,
                    const int*   __restrict__ labels,
                    const float* __restrict__ sq,
                    float* __restrict__ pos2,
                    float* __restrict__ neg2) {
  __shared__ _Float16 sHi[CTILE * LDS_STRIDE];
  __shared__ _Float16 sLo[CTILE * LDS_STRIDE];
  __shared__ int      sLab[CTILE];
  __shared__ float    sSq[CTILE];

  const int t    = threadIdx.x;
  const int lane = t & 31;
  const int w    = t >> 5;        // wave id 0..7
  const int g    = lane >> 4;     // 16-lane half
  const int l16  = lane & 15;

  const int rowBase  = blockIdx.y * ROWS_PER_WG + w * 16;
  const int colBase0 = blockIdx.x * COLS_PER_SPLIT;

  // ---- Preload A tile (16 rows x 128 K) as hi/lo f16 in VGPRs ----
  // ISA A layout (16-bit 16x32): lane half g -> elems 0..7 = K[8g..8g+7],
  // elems 8..15 = K[16+8g..23+8g] within each K=32 chunk.
  v16h aHi[4], aLo[4];
  {
    const float* arow = E + (size_t)(rowBase + l16) * TL_D;
#pragma unroll
    for (int c = 0; c < 4; ++c) {
      const int kb = 32 * c + 8 * g;
      const float4* p0 = (const float4*)(arow + kb);
      const float4* p1 = (const float4*)(arow + kb + 16);
      float4 v0 = p0[0], v1 = p0[1], v2 = p1[0], v3 = p1[1];
      float xs[16] = {v0.x, v0.y, v0.z, v0.w, v1.x, v1.y, v1.z, v1.w,
                      v2.x, v2.y, v2.z, v2.w, v3.x, v3.y, v3.z, v3.w};
#pragma unroll
      for (int e = 0; e < 16; ++e) {
        float x = xs[e];
        _Float16 h = (_Float16)x;
        aHi[c][e] = h;
        aLo[c][e] = (_Float16)(x - (float)h);
      }
    }
  }

  // Row-side metadata in the f32 C/D layout: VGPR r, half g -> row r + 8g.
  int rowG[8]; int labR[8]; float sqR[8];
#pragma unroll
  for (int r = 0; r < 8; ++r) {
    int m = rowBase + r + 8 * g;
    rowG[r] = m;
    labR[r] = labels[m];
    sqR[r]  = sq[m];
  }

  const float INF = __builtin_inff();
  float p2[8], n2[8];
#pragma unroll
  for (int r = 0; r < 8; ++r) { p2[r] = -INF; n2[r] = INF; }

  for (int ct = 0; ct < COLS_PER_SPLIT / CTILE; ++ct) {
    const int colBase = colBase0 + ct * CTILE;
    __syncthreads();
    // ---- Cooperative fill: 32 cols x 128 K fp32 -> hi/lo f16 LDS tiles ----
    // thread t handles 16 contiguous K of one column.
    {
      const int col = t >> 3;
      const int k0  = (t & 7) * 16;
      const float4* gp = (const float4*)(E + (size_t)(colBase + col) * TL_D + k0);
      float4 x0 = gp[0], x1 = gp[1], x2 = gp[2], x3 = gp[3];
      float xs[16] = {x0.x, x0.y, x0.z, x0.w, x1.x, x1.y, x1.z, x1.w,
                      x2.x, x2.y, x2.z, x2.w, x3.x, x3.y, x3.z, x3.w};
      v8h hv0, lv0, hv1, lv1;
#pragma unroll
      for (int e = 0; e < 8; ++e) {
        float x = xs[e];
        _Float16 h = (_Float16)x;
        hv0[e] = h;
        lv0[e] = (_Float16)(x - (float)h);
        float y = xs[e + 8];
        _Float16 h2 = (_Float16)y;
        hv1[e] = h2;
        lv1[e] = (_Float16)(y - (float)h2);
      }
      *(v8h*)&sHi[col * LDS_STRIDE + k0]     = hv0;
      *(v8h*)&sHi[col * LDS_STRIDE + k0 + 8] = hv1;
      *(v8h*)&sLo[col * LDS_STRIDE + k0]     = lv0;
      *(v8h*)&sLo[col * LDS_STRIDE + k0 + 8] = lv1;
      if (t < CTILE) { sLab[t] = labels[colBase + t]; sSq[t] = sq[colBase + t]; }
      // prefetch next column tile into cache
      if (ct + 1 < COLS_PER_SPLIT / CTILE)
        __builtin_prefetch(E + (size_t)(colBase + CTILE + col) * TL_D + k0, 0, 3);
    }
    __syncthreads();

    // ---- Two 16x16x128 Gram subtiles: 3 f16 WMMAs per K=32 chunk each ----
    v8f acc[2];
    acc[0] = (v8f){};
    acc[1] = (v8f){};
#pragma unroll
    for (int s = 0; s < 2; ++s) {
      const int brow = (s * 16 + l16) * LDS_STRIDE;
#pragma unroll
      for (int c = 0; c < 4; ++c) {
        // ISA B layout (16-bit 32x16): lane half g -> elems 0..7 = K[8g..8g+7], elems 8..15 = K[16+8g..23+8g]
        const int kb = 32 * c + 8 * g;
        v8h bh0 = *(const v8h*)&sHi[brow + kb];
        v8h bh1 = *(const v8h*)&sHi[brow + kb + 16];
        v8h bl0 = *(const v8h*)&sLo[brow + kb];
        v8h bl1 = *(const v8h*)&sLo[brow + kb + 16];
        v16h bHi = __builtin_shufflevector(bh0, bh1, 0,1,2,3,4,5,6,7,8,9,10,11,12,13,14,15);
        v16h bLo = __builtin_shufflevector(bl0, bl1, 0,1,2,3,4,5,6,7,8,9,10,11,12,13,14,15);
        acc[s] = __builtin_amdgcn_wmma_f32_16x16x32_f16(false, aHi[c], false, bHi,
                                                        (short)0, acc[s], false, false);
        acc[s] = __builtin_amdgcn_wmma_f32_16x16x32_f16(false, aHi[c], false, bLo,
                                                        (short)0, acc[s], false, false);
        acc[s] = __builtin_amdgcn_wmma_f32_16x16x32_f16(false, aLo[c], false, bHi,
                                                        (short)0, acc[s], false, false);
      }
    }

    // ---- Branch-free masked running max/min on raw squared distance q ----
    // Masked-out candidates are +/-inf (identity for min/max): pure
    // v_cndmask + v_max/min_num_f32, no exec divergence, co-executes with XDL.
#pragma unroll
    for (int s = 0; s < 2; ++s) {
      const int   colJ = colBase + s * 16 + l16;
      const int   labC = sLab[s * 16 + l16];
      const float sqC  = sSq[s * 16 + l16];
#pragma unroll
      for (int r = 0; r < 8; ++r) {
        float q    = __builtin_fmaf(-2.0f, acc[s][r], sqR[r] + sqC);
        bool  same = (labC == labR[r]);
        bool  self = (colJ == rowG[r]);
        float qp   = (same && !self) ? q : -INF;  // pos candidate
        float qn   = same ? INF : q;              // neg candidate
        p2[r] = fmaxf(p2[r], qp);
        n2[r] = fminf(n2[r], qn);
      }
    }
  }

  // ---- Butterfly reduce across the 16 lanes holding each row's columns ----
#pragma unroll
  for (int r = 0; r < 8; ++r) {
#pragma unroll
    for (int m = 1; m < 16; m <<= 1) {
      p2[r] = fmaxf(p2[r], __shfl_xor(p2[r], m, 32));
      n2[r] = fminf(n2[r], __shfl_xor(n2[r], m, 32));
    }
  }
  // Combine partial results across the column-split WGs (native f32 atomics).
  if (l16 == 0) {
#pragma unroll
    for (int r = 0; r < 8; ++r) {
      atomicMax(&pos2[rowG[r]], p2[r]);
      atomicMin(&neg2[rowG[r]], n2[r]);
    }
  }
}

// ---------------------------------------------------------------------------
// Kernel 3: per-row loss (deferred relu/eps/sqrt), mean over valid rows.
// ---------------------------------------------------------------------------
__global__ void triplet_final_kernel(const float* __restrict__ pos2,
                                     const float* __restrict__ neg2,
                                     float* __restrict__ out) {
  __shared__ float sSum[256];
  __shared__ int   sCnt[256];
  const int t = threadIdx.x;
  float s = 0.0f;
  int   c = 0;
  for (int i = t; i < TL_B; i += 256) {
    float p = pos2[i], n = neg2[i];
    bool valid = (p != -__builtin_inff()) & (n != __builtin_inff());
    float pd = sqrtf(fmaxf(p, 0.0f) + TL_EPS);
    float nd = sqrtf(fmaxf(n, 0.0f) + TL_EPS);
    float l  = fmaxf(pd - nd + TL_MARGIN, 0.0f);
    s += valid ? l : 0.0f;
    c += valid ? 1 : 0;
  }
  sSum[t] = s;
  sCnt[t] = c;
  __syncthreads();
  for (int off = 128; off > 0; off >>= 1) {
    if (t < off) { sSum[t] += sSum[t + off]; sCnt[t] += sCnt[t + off]; }
    __syncthreads();
  }
  if (t == 0) {
    int n = sCnt[0];
    out[0] = (n > 0) ? (sSum[0] / (float)n) : 0.0f;
    out[1] = (float)n;
  }
}

// ---------------------------------------------------------------------------
extern "C" void kernel_launch(void* const* d_in, const int* in_sizes, int n_in,
                              void* d_out, int out_size, void* d_ws, size_t ws_size,
                              hipStream_t stream) {
  const float* E      = (const float*)d_in[0];   // (8192, 128) fp32
  const int*   labels = (const int*)d_in[1];     // (8192,) int32
  float*       out    = (float*)d_out;           // [loss, n_valid]

  float* ws   = (float*)d_ws;                    // 3 * 8192 floats = 96 KB
  float* sq   = ws;
  float* pos2 = ws + TL_B;
  float* neg2 = ws + 2 * TL_B;

  triplet_init_kernel<<<TL_B / 256, 256, 0, stream>>>(E, sq, pos2, neg2);

  dim3 grid(COL_SPLITS, TL_B / ROWS_PER_WG);     // 8 x 64
  triplet_main_kernel<<<grid, 256, 0, stream>>>(E, labels, sq, pos2, neg2);

  triplet_final_kernel<<<1, 256, 0, stream>>>(pos2, neg2, out);
}
